// GIBGCN_75986561401517
// MI455X (gfx1250) — matrix operand order repaired and checked by reading, not compiled, not run on hardware
//
#include <hip/hip_runtime.h>

// ---------------------------------------------------------------------------
// Problem sizes (fixed by the reference)
// ---------------------------------------------------------------------------
#define N_NODES 100000
#define N_EDGES 3200000
#define NFEAT   500
#define LAT     32
#define NCLS    16

// GEMM1 tiling
#define TILE_M    64          // rows per block (4 waves x 16)
#define ROW_PITCH 516         // dwords per LDS row: 500 data + 16 zero pad; 516 % 64 == 4
#define KTILES    16          // ceil(500/32): K padded to 512 with zeros
#define W1PACK_N  (KTILES * 2 * 32 * 16)   // 16384 bf16
#define W2PACK_N  (32 * 16)                //   512 bf16

typedef __bf16 bf16;
typedef __attribute__((ext_vector_type(16))) bf16         v16bf;
typedef __attribute__((ext_vector_type(8)))  float        v8f;
typedef __attribute__((ext_vector_type(4)))  unsigned int v4u;
typedef __attribute__((ext_vector_type(8)))  int          v8i;
typedef __attribute__((ext_vector_type(4)))  int          v4i;

#if __has_builtin(__builtin_amdgcn_tensor_load_to_lds)
#define HAVE_TDM 1
#else
#define HAVE_TDM 0
#endif

// ---------------------------------------------------------------------------
// Pack W1 (500x32) and W2 (32x16) into bf16 WMMA B-fragment layout (ISA 7.12.2):
// per lane l, element i of tile (t, ntile nt):  K = t*32 + 8*(l>>4) + (i&7) + 16*(i>>3),
// N = nt*16 + (l&15). Fragment for (t,nt,l) is 16 contiguous bf16 (32B).
// ---------------------------------------------------------------------------
__global__ void wpack_kernel(const float* __restrict__ W1,
                             const float* __restrict__ W2,
                             bf16* __restrict__ wp1, bf16* __restrict__ wp2) {
  const int j = blockIdx.x * blockDim.x + threadIdx.x;
  if (j < W1PACK_N) {
    const int i  = j & 15;
    const int l  = (j >> 4) & 31;
    const int nt = (j >> 9) & 1;
    const int t  = j >> 10;
    const int k  = t * 32 + 8 * (l >> 4) + (i & 7) + 16 * (i >> 3);
    wp1[j] = (bf16)((k < NFEAT) ? W1[k * LAT + nt * 16 + (l & 15)] : 0.0f);
  } else if (j < W1PACK_N + W2PACK_N) {
    const int jj = j - W1PACK_N;
    const int i  = jj & 15;
    const int l  = jj >> 4;
    const int k  = (i & 7) + 16 * (i >> 3) + 8 * (l >> 4);  // 0..31
    wp2[jj] = (bf16)W2[k * NCLS + (l & 15)];
  }
}

// ---------------------------------------------------------------------------
// GEMM1: h1[N_NODES x 32] = x[N_NODES x 500] @ W1[500 x 32]   (bf16 WMMA)
// x tile (64 rows x 516 dwords, zero-padded by TDM OOB rule) staged in LDS by
// the Tensor Data Mover; B fragments pre-packed in global (L2-resident).
// ---------------------------------------------------------------------------
__global__ __launch_bounds__(128) void gemm1_kernel(
    const float* __restrict__ x, const bf16* __restrict__ wp1,
    float* __restrict__ h1) {
  extern __shared__ float xs[];            // TILE_M * ROW_PITCH floats
  const int tid   = threadIdx.x;
  const int lane  = tid & 31;
  const int wave  = tid >> 5;              // 0..3
  const int m_blk = blockIdx.x * TILE_M;

#if HAVE_TDM
  if (wave == 0) {
    // ---- Tensor DMA: 2D tile, tile_dim0=516 > tensor_dim0=500 -> K pad
    // zero-filled; rows beyond tensor_dim1 zero-filled (tail blocks). ----
    const unsigned long long gaddr =
        (unsigned long long)(const void*)(x + (size_t)m_blk * NFEAT);
    const unsigned int ldsa = (unsigned int)(unsigned long long)(void*)xs;
    const int dim1 = N_NODES - m_blk;      // remaining rows in tensor
    v4u g0;
    g0.x = 1u;                              // count=1, user descriptor
    g0.y = ldsa;                            // lds_addr
    g0.z = (unsigned int)gaddr;             // global_addr[31:0]
    g0.w = ((unsigned int)(gaddr >> 32) & 0x01FFFFFFu) | 0x80000000u; // addr[56:32] | type=2
    v8i g1;
    g1[0] = 0x00020000;                     // data_size = 2 (4 bytes)
    g1[1] = (int)((unsigned)NFEAT << 16);   // tensor_dim0[15:0] @ bits63:48
    g1[2] = (int)(((unsigned)dim1 & 0xFFFFu) << 16);        // tensor_dim1[15:0]
    g1[3] = (int)(((unsigned)dim1 >> 16) | (516u << 16));   // tensor_dim1[31:16] | tile_dim0
    g1[4] = TILE_M;                         // tile_dim1 = 64, tile_dim2 = 0
    g1[5] = NFEAT;                          // tensor_dim0_stride[31:0] = 500
    g1[6] = 0;
    g1[7] = 0;
    v4i gz = {0, 0, 0, 0};
#if __has_include(<hip/amd_detail/amd_gfx1250_TDM.h>)
    v8i gz8 = {0, 0, 0, 0, 0, 0, 0, 0};
    __builtin_amdgcn_tensor_load_to_lds(g0, g1, gz, gz, gz8, 0);
#else
    __builtin_amdgcn_tensor_load_to_lds(g0, g1, gz, gz, 0);
#endif
    __builtin_amdgcn_s_wait_tensorcnt(0);
  }
#else
  // Fallback: cooperative copy with zero padding.
  for (int r = wave; r < TILE_M; r += 4) {
    const int grow = m_blk + r;
    const float* src = x + (size_t)grow * NFEAT;
    float* dst = xs + r * ROW_PITCH;
    for (int c = lane * 4; c < ROW_PITCH; c += 128) {
      float4 v = {0.0f, 0.0f, 0.0f, 0.0f};
      if (grow < N_NODES && c < NFEAT) v = *(const float4*)(src + c);
      *(float4*)(dst + c) = v;
    }
  }
#endif
  __syncthreads();

  const int half = lane >> 4;              // 0: lanes 0-15, 1: lanes 16-31
  const int col  = lane & 15;
  const float* arow = xs + (wave * 16 + col) * ROW_PITCH;   // this lane's A row

  v8f c0 = {};  // columns 0..15
  v8f c1 = {};  // columns 16..31
#pragma unroll
  for (int t = 0; t < KTILES; ++t) {
    const int k0 = t * 32;
    // A fragment: two contiguous 8-float runs per lane -> 4x ds_load_b128.
    const float4 f0 = *(const float4*)(arow + k0 + 8 * half);
    const float4 f1 = *(const float4*)(arow + k0 + 8 * half + 4);
    const float4 f2 = *(const float4*)(arow + k0 + 16 + 8 * half);
    const float4 f3 = *(const float4*)(arow + k0 + 16 + 8 * half + 4);
    v16bf a;
    a[0]  = (bf16)f0.x; a[1]  = (bf16)f0.y; a[2]  = (bf16)f0.z; a[3]  = (bf16)f0.w;
    a[4]  = (bf16)f1.x; a[5]  = (bf16)f1.y; a[6]  = (bf16)f1.z; a[7]  = (bf16)f1.w;
    a[8]  = (bf16)f2.x; a[9]  = (bf16)f2.y; a[10] = (bf16)f2.z; a[11] = (bf16)f2.w;
    a[12] = (bf16)f3.x; a[13] = (bf16)f3.y; a[14] = (bf16)f3.z; a[15] = (bf16)f3.w;
    // B fragments: 32B contiguous per lane from the packed table (L2-hot).
    const v16bf b0 = *(const v16bf*)(wp1 + ((size_t)(t * 2 + 0) * 32 + lane) * 16);
    const v16bf b1 = *(const v16bf*)(wp1 + ((size_t)(t * 2 + 1) * 32 + lane) * 16);
    c0 = __builtin_amdgcn_wmma_f32_16x16x32_bf16(false, a, false, b0,
                                                 (short)0, c0, false, false);
    c1 = __builtin_amdgcn_wmma_f32_16x16x32_bf16(false, a, false, b1,
                                                 (short)0, c1, false, false);
  }

  // C layout: VGPR r -> M = m_blk + wave*16 + r + 8*half, N = col.
  const int m0 = m_blk + wave * 16;
#pragma unroll
  for (int r = 0; r < 8; ++r) {
    const int row = m0 + r + 8 * half;
    if (row < N_NODES) {
      h1[(size_t)row * LAT + col]      = c0[r];
      h1[(size_t)row * LAT + 16 + col] = c1[r];
    }
  }
}

// ---------------------------------------------------------------------------
// GEMM2: h2[N_NODES x 16] = out1[N_NODES x 32] @ W2[32 x 16]
// K = 32 -> one v_wmma per 16x16 tile. Rows clamped (never predicated):
// clamped rows only feed output rows that are not stored.
// ---------------------------------------------------------------------------
__global__ __launch_bounds__(256) void gemm2_kernel(
    const float* __restrict__ in, const bf16* __restrict__ wp2,
    float* __restrict__ h2) {
  const int lane = threadIdx.x & 31;
  const int wave = threadIdx.x >> 5;
  const int m0   = (blockIdx.x * 8 + wave) * 16;
  if (m0 >= N_NODES) return;               // uniform per wave

  const int half = lane >> 4;
  const int col  = lane & 15;
  int mrow = m0 + col;
  if (mrow > N_NODES - 1) mrow = N_NODES - 1;      // clamp, no divergence
  const float* irow = in + (size_t)mrow * LAT;

  const float4 f0 = *(const float4*)(irow + 8 * half);
  const float4 f1 = *(const float4*)(irow + 8 * half + 4);
  const float4 f2 = *(const float4*)(irow + 16 + 8 * half);
  const float4 f3 = *(const float4*)(irow + 16 + 8 * half + 4);
  v16bf a;
  a[0]  = (bf16)f0.x; a[1]  = (bf16)f0.y; a[2]  = (bf16)f0.z; a[3]  = (bf16)f0.w;
  a[4]  = (bf16)f1.x; a[5]  = (bf16)f1.y; a[6]  = (bf16)f1.z; a[7]  = (bf16)f1.w;
  a[8]  = (bf16)f2.x; a[9]  = (bf16)f2.y; a[10] = (bf16)f2.z; a[11] = (bf16)f2.w;
  a[12] = (bf16)f3.x; a[13] = (bf16)f3.y; a[14] = (bf16)f3.z; a[15] = (bf16)f3.w;
  const v16bf b = *(const v16bf*)(wp2 + (size_t)lane * 16);

  v8f c = {};
  c = __builtin_amdgcn_wmma_f32_16x16x32_bf16(false, a, false, b,
                                              (short)0, c, false, false);
#pragma unroll
  for (int r = 0; r < 8; ++r) {
    const int row = m0 + r + 8 * half;
    if (row < N_NODES) h2[(size_t)row * NCLS + col] = c[r];
  }
}

// ---------------------------------------------------------------------------
// Bias broadcast init: out[i] = b[i & mask]   (C is a power of two)
// ---------------------------------------------------------------------------
__global__ void bias_init_kernel(float* __restrict__ out,
                                 const float* __restrict__ b,
                                 int mask, long total) {
  const long i = (long)blockIdx.x * blockDim.x + threadIdx.x;
  if (i < total) out[i] = b[(int)i & mask];
}

// ---------------------------------------------------------------------------
// Edge scatter, layer 1 (32 channels): 8 lanes per edge, 4 channels per lane.
// h1[src] gather is a coalesced 128B row; f32 atomics land in L2 (12.8 MB table).
// ---------------------------------------------------------------------------
__global__ __launch_bounds__(256) void scatter1_kernel(
    const float* __restrict__ h, const int* __restrict__ ei,
    const float* __restrict__ ew, float* __restrict__ out) {
  const long tid = (long)blockIdx.x * blockDim.x + threadIdx.x;
  if (tid >= (long)N_EDGES * 8) return;
  const int e = (int)(tid >> 3);
  const int g = (int)(tid & 7);
  __builtin_prefetch(ei + e + 2048, 0, 0);
  const int src = ei[e];
  const int dst = ei[N_EDGES + e];
  const float w = ew[e];
  const float4 v = *(const float4*)(h + (size_t)src * LAT + g * 4);
  float* o = out + (size_t)dst * LAT + g * 4;
  atomicAdd(o + 0, v.x * w);
  atomicAdd(o + 1, v.y * w);
  atomicAdd(o + 2, v.z * w);
  atomicAdd(o + 3, v.w * w);
}

// ---------------------------------------------------------------------------
// Edge scatter, layer 2 (16 channels): 4 lanes per edge, 4 channels per lane.
// ---------------------------------------------------------------------------
__global__ __launch_bounds__(256) void scatter2_kernel(
    const float* __restrict__ h, const int* __restrict__ ei,
    const float* __restrict__ ew, float* __restrict__ out) {
  const long tid = (long)blockIdx.x * blockDim.x + threadIdx.x;
  if (tid >= (long)N_EDGES * 4) return;
  const int e = (int)(tid >> 2);
  const int g = (int)(tid & 3);
  const int src = ei[e];
  const int dst = ei[N_EDGES + e];
  const float w = ew[e];
  const float4 v = *(const float4*)(h + (size_t)src * NCLS + g * 4);
  float* o = out + (size_t)dst * NCLS + g * 4;
  atomicAdd(o + 0, v.x * w);
  atomicAdd(o + 1, v.y * w);
  atomicAdd(o + 2, v.z * w);
  atomicAdd(o + 3, v.w * w);
}

// ---------------------------------------------------------------------------
// Diagonal-Gaussian KL (vectorized):
// mean = out[:, :C/2], std = softplus(out[:, C/2:]) + 1e-10
// kl = sum(-log(std) + 0.5*(std^2 + mean^2 - 1))
// ---------------------------------------------------------------------------
__device__ __forceinline__ float kl_term(float m, float v) {
  const float sp = (v > 20.0f) ? v : log1pf(expf(v));
  const float s  = sp + 1e-10f;
  return -logf(s) + 0.5f * (s * s + m * m - 1.0f);
}

template <int C>
__global__ void ixz_kernel(const float* __restrict__ out,
                           float* __restrict__ ixz, int n) {
  const int i = blockIdx.x * blockDim.x + threadIdx.x;
  if (i >= n) return;
  constexpr int HALF = C / 2;
  const float* row = out + (size_t)i * C;
  float kl = 0.0f;
#pragma unroll
  for (int c = 0; c < HALF; c += 4) {
    const float4 m = *(const float4*)(row + c);
    const float4 v = *(const float4*)(row + HALF + c);
    kl += kl_term(m.x, v.x) + kl_term(m.y, v.y) +
          kl_term(m.z, v.z) + kl_term(m.w, v.w);
  }
  ixz[i] = kl;
}

__global__ void skl_kernel(float* __restrict__ p) {
  if (blockIdx.x == 0 && threadIdx.x == 0) *p = 0.0f;
}

// ---------------------------------------------------------------------------
// Host launcher
// d_in: x, edge_index, edge_weight, W1, b1, W2, b2
// d_out: [out2 (100000x16) | out1 (100000x32) | ixz1 | ixz2 | skl]
// d_ws:  [h1 (100000x32 f32) | h2 (100000x16 f32) | wp1 (16384 bf16) | wp2 (512 bf16)]
// ---------------------------------------------------------------------------
extern "C" void kernel_launch(void* const* d_in, const int* in_sizes, int n_in,
                              void* d_out, int out_size, void* d_ws, size_t ws_size,
                              hipStream_t stream) {
  const float* x  = (const float*)d_in[0];
  const int*   ei = (const int*)  d_in[1];
  const float* ew = (const float*)d_in[2];
  const float* W1 = (const float*)d_in[3];
  const float* b1 = (const float*)d_in[4];
  const float* W2 = (const float*)d_in[5];
  const float* b2 = (const float*)d_in[6];

  float* out  = (float*)d_out;
  float* out2 = out;
  float* out1 = out2 + (size_t)N_NODES * NCLS;
  float* ixz1 = out1 + (size_t)N_NODES * LAT;
  float* ixz2 = ixz1 + N_NODES;
  float* skl  = ixz2 + N_NODES;

  float* h1  = (float*)d_ws;                           // 12.8 MB
  float* h2  = h1 + (size_t)N_NODES * LAT;             //  6.4 MB
  bf16*  wp1 = (bf16*)(h2 + (size_t)N_NODES * NCLS);   // 32 KB
  bf16*  wp2 = wp1 + W1PACK_N;                         //  1 KB

  // Pack weights into WMMA B-fragment layout (runs once per launch).
  wpack_kernel<<<(W1PACK_N + W2PACK_N + 255) / 256, 256, 0, stream>>>(
      W1, W2, wp1, wp2);

  // Layer 1
  const int g1_grid = (N_NODES + TILE_M - 1) / TILE_M;          // 1563
  const size_t g1_smem = (size_t)TILE_M * ROW_PITCH * sizeof(float);  // 132 KB
  gemm1_kernel<<<g1_grid, 128, g1_smem, stream>>>(x, wp1, h1);
  bias_init_kernel<<<(N_NODES * LAT + 255) / 256, 256, 0, stream>>>(
      out1, b1, LAT - 1, (long)N_NODES * LAT);
  scatter1_kernel<<<(int)(((long)N_EDGES * 8 + 255) / 256), 256, 0, stream>>>(
      h1, ei, ew, out1);
  ixz_kernel<LAT><<<(N_NODES + 255) / 256, 256, 0, stream>>>(out1, ixz1, N_NODES);

  // Layer 2
  const int g2_grid = (N_NODES + 127) / 128;                    // 782
  gemm2_kernel<<<g2_grid, 256, 0, stream>>>(out1, wp2, h2);
  bias_init_kernel<<<(N_NODES * NCLS + 255) / 256, 256, 0, stream>>>(
      out2, b2, NCLS - 1, (long)N_NODES * NCLS);
  scatter2_kernel<<<(int)(((long)N_EDGES * 4 + 255) / 256), 256, 0, stream>>>(
      h2, ei, ew, out2);
  ixz_kernel<NCLS><<<(N_NODES + 255) / 256, 256, 0, stream>>>(out2, ixz2, N_NODES);

  skl_kernel<<<1, 32, 0, stream>>>(skl);
}
